// VectorQuantizer_9311489098058
// MI455X (gfx1250) — matrix operand order repaired and checked by reading, not compiled, not run on hardware
//
#include <hip/hip_runtime.h>
#include <hip/hip_bf16.h>

typedef __attribute__((ext_vector_type(16))) __bf16 v16bf;
typedef __attribute__((ext_vector_type(8)))  float  v8f;
typedef __attribute__((ext_vector_type(4)))  int    nint4;   // native 4xint vector

#define D_DIM   256
#define HW      4096
#define KCODES  2048
#define N_TOT   65536
#define ROWS_PB 64          // z rows per block (4 waves x 16)
#define THREADS 128
#define CHUNK   32          // codebook rows per LDS chunk (double-buffered)
#define NCHUNKS (KCODES / CHUNK)
#define LDS_STRIDE 132      // dwords per row: 128 data + 4 pad (16B aligned)
#define BUF_DW  (CHUNK * LDS_STRIDE)   // 4224 dwords per chunk buffer

// ws layout (dword units): [0]=sqsum(f32), [64..2111]=cnorm(f32),
// [2112..4159]=counts(i32), [4160..]=codebook bf16 packed (2048*128 dwords)
#define WS_SQSUM 0
#define WS_CNORM 64
#define WS_CNT   (64 + KCODES)
#define WS_CBBF  (64 + KCODES + KCODES)

// ---- CDNA5 async global->LDS path (probe-guarded; sync fallback) ----
#if defined(__has_builtin)
#  if __has_builtin(__builtin_amdgcn_global_load_async_to_lds_b128)
#    define HAVE_ASYNC 1
#  endif
#endif
#ifndef HAVE_ASYNC
#  define HAVE_ASYNC 0
#endif

#if HAVE_ASYNC
// param0 = addrspace(1) pointer to native int4 vector (from hipcc diagnostics)
typedef __attribute__((address_space(1))) nint4* gasp;
typedef __attribute__((address_space(3))) nint4* lasp;
__device__ __forceinline__ void wait_async0() {
#if defined(__has_builtin) && __has_builtin(__builtin_amdgcn_s_wait_asynccnt)
  __builtin_amdgcn_s_wait_asynccnt(0);
#else
  asm volatile("s_wait_asynccnt 0x0" ::: "memory");
#endif
}
#else
__device__ __forceinline__ void wait_async0() {}
#endif

__device__ __forceinline__ unsigned int f2bf_pack(float lo, float hi) {
  unsigned int a = __float_as_uint(lo);
  unsigned int b = __float_as_uint(hi);
  a += 0x7fffu + ((a >> 16) & 1u);   // round-to-nearest-even
  b += 0x7fffu + ((b >> 16) & 1u);
  return (a >> 16) | (b & 0xffff0000u);
}

// ---------------- Kernel 1: bf16 codebook + norms + zero accumulators ----------------
__global__ void vq_prep(const float* __restrict__ cb, float* wsf, int* wsi,
                        unsigned int* __restrict__ cbbf) {
  int g = blockIdx.x * blockDim.x + threadIdx.x;   // 0..2047 (one per code)
  if (g == 0) wsf[WS_SQSUM] = 0.f;
  const float* row = cb + g * D_DIM;
  unsigned int* orow = cbbf + g * 128;
  float s = 0.f;
  #pragma unroll 4
  for (int dw = 0; dw < 128; ++dw) {
    float f0 = row[2 * dw], f1 = row[2 * dw + 1];
    s += f0 * f0 + f1 * f1;
    orow[dw] = f2bf_pack(f0, f1);
  }
  wsf[WS_CNORM + g] = s;
  wsi[WS_CNT + g] = 0;
}

// ---------------- Kernel 2: fused distance-GEMM + argmin + outputs ----------------
__global__ __launch_bounds__(THREADS) void vq_main(
    const float* __restrict__ z, const float* __restrict__ cb,
    const unsigned int* __restrict__ cbbf,
    float* __restrict__ q, float* wsf, int* wsi) {
  // z tile (64x132 dw) aliases the two 32x132 chunk buffers
  __shared__ unsigned int smem[2 * BUF_DW];
  __shared__ float s_cnorm[2][CHUNK];
  __shared__ int   s_idx[ROWS_PB];
  __shared__ float s_acc;

  const int tid  = threadIdx.x;
  const int lane = tid & 31;
  const int wave = tid >> 5;          // 0..3
  const int M    = lane & 15;         // row (A) / column (B,C) within tile
  const int kh   = lane >> 4;         // lane half
  const long long nBase = (long long)blockIdx.x * ROWS_PB;

  if (tid == 0) s_acc = 0.f;

  // ---- stage 64x256 z tile into LDS as packed bf16 (coalesced over hw) ----
  for (int i = 0; i < 64; ++i) {
    int flat = i * THREADS + tid;          // 0..8191
    int r  = flat & 63;                    // local row (fastest -> coalesced)
    int dw = flat >> 6;                    // dword index 0..127 (pair of dims)
    long long n  = nBase + r;
    long long b  = n >> 12;
    long long hw = n & 4095;
    long long addr = b * ((long long)D_DIM * HW) + (long long)(dw * 2) * HW + hw;
    smem[r * LDS_STRIDE + dw] = f2bf_pack(z[addr], z[addr + HW]);
  }
  __syncthreads();

  // ---- each wave loads its whole 16x256 A operand into registers ----
  // ISA 16-bit A 16x32 layout: lane M holds K = kh*8 + {0..7} (VGPR0..3) and
  // K = 16 + kh*8 + {0..7} (VGPR4..7)  => dwords base, base+8 (4 each)
  union Frag { nint4 h[2]; v16bf v; };
  Frag afrag[8];
  {
    const int rowBase = (wave * 16 + M) * LDS_STRIDE;
    #pragma unroll
    for (int s = 0; s < 8; ++s) {
      int base = rowBase + s * 16 + kh * 4;
      afrag[s].h[0] = *reinterpret_cast<const nint4*>(&smem[base]);
      afrag[s].h[1] = *reinterpret_cast<const nint4*>(&smem[base + 8]);
    }
  }
  __syncthreads();   // all waves done reading z tile: buffers reusable

  float rmax[8];
  int   ridx[8];
  #pragma unroll
  for (int j = 0; j < 8; ++j) { rmax[j] = -3.0e38f; ridx[j] = 0; }

  // ---- prologue: stage chunk 0 (chunk c lives in buffer (c+1)&1) ----
  if (tid < CHUNK) s_cnorm[0][tid] = wsf[WS_CNORM + tid];
  {
    unsigned int* bufp = smem + 1 * BUF_DW;
    #pragma unroll
    for (int i = 0; i < 8; ++i) {
      int flat = i * THREADS + tid;       // 0..1023
      int seg = flat & 31;                // 16B segment (fastest -> coalesced)
      int c   = flat >> 5;                // 0..31 local code
      int gdw = c * 128 + seg * 4;
#if HAVE_ASYNC
      __builtin_amdgcn_global_load_async_to_lds_b128(
          (gasp)(cbbf + gdw),
          (lasp)(bufp + c * LDS_STRIDE + seg * 4), 0, 0);
#else
      nint4 v = *reinterpret_cast<const nint4*>(cbbf + gdw);
      *reinterpret_cast<nint4*>(bufp + c * LDS_STRIDE + seg * 4) = v;
#endif
    }
  }
  wait_async0();
  __syncthreads();

  // ---- K loop: double-buffered, async staging overlapped with WMMA ----
  for (int ch = 0; ch < NCHUNKS; ++ch) {
    // fire chunk ch+1 into the other buffer
    if (ch + 1 < NCHUNKS) {
      unsigned int* bufp = smem + (ch & 1) * BUF_DW;   // buffer ((ch+1)+1)&1
      const int kbase = (ch + 1) * CHUNK;
      #pragma unroll
      for (int i = 0; i < 8; ++i) {
        int flat = i * THREADS + tid;
        int seg = flat & 31;
        int c   = flat >> 5;
        int gdw = (kbase + c) * 128 + seg * 4;
#if HAVE_ASYNC
        __builtin_amdgcn_global_load_async_to_lds_b128(
            (gasp)(cbbf + gdw),
            (lasp)(bufp + c * LDS_STRIDE + seg * 4), 0, 0);
#else
        nint4 v = *reinterpret_cast<const nint4*>(cbbf + gdw);
        *reinterpret_cast<nint4*>(bufp + c * LDS_STRIDE + seg * 4) = v;
#endif
      }
      if (tid < CHUNK)
        s_cnorm[(ch + 1) & 1][tid] = wsf[WS_CNORM + kbase + tid];
    }

    // ---- compute chunk ch from buffer (ch+1)&1 ----
    // dual-tile interleave (2 independent WMMA chains) + B loads pipelined
    // one K-step ahead so ds latency hides under two WMMAs.
    {
      const unsigned int* cbuf = smem + (((ch + 1) & 1)) * BUF_DW;
      const int r0 = (M)      * LDS_STRIDE;   // tile 0: codes 0..15
      const int r1 = (16 + M) * LDS_STRIDE;   // tile 1: codes 16..31
      const int ko = kh * 4;
      v8f acc0 = {0.f, 0.f, 0.f, 0.f, 0.f, 0.f, 0.f, 0.f};
      v8f acc1 = {0.f, 0.f, 0.f, 0.f, 0.f, 0.f, 0.f, 0.f};
      Frag b0, b1;
      b0.h[0] = *reinterpret_cast<const nint4*>(&cbuf[r0 + ko]);
      b0.h[1] = *reinterpret_cast<const nint4*>(&cbuf[r0 + ko + 8]);
      b1.h[0] = *reinterpret_cast<const nint4*>(&cbuf[r1 + ko]);
      b1.h[1] = *reinterpret_cast<const nint4*>(&cbuf[r1 + ko + 8]);
      #pragma unroll
      for (int s = 0; s < 8; ++s) {          // D = 256 in 8 steps of 32
        Frag n0, n1;
        if (s < 7) {
          int nb0 = r0 + (s + 1) * 16 + ko;
          int nb1 = r1 + (s + 1) * 16 + ko;
          n0.h[0] = *reinterpret_cast<const nint4*>(&cbuf[nb0]);
          n0.h[1] = *reinterpret_cast<const nint4*>(&cbuf[nb0 + 8]);
          n1.h[0] = *reinterpret_cast<const nint4*>(&cbuf[nb1]);
          n1.h[1] = *reinterpret_cast<const nint4*>(&cbuf[nb1 + 8]);
        }
        acc0 = __builtin_amdgcn_wmma_f32_16x16x32_bf16(
                   false, afrag[s].v, false, b0.v, (short)0, acc0, false, false);
        acc1 = __builtin_amdgcn_wmma_f32_16x16x32_bf16(
                   false, afrag[s].v, false, b1.v, (short)0, acc1, false, false);
        if (s < 7) { b0 = n0; b1 = n1; }
      }
      // score = 2*z.c - ||c||^2 ; C layout: lane = column N, slot j = row M(+8*kh)
      float cn0 = s_cnorm[ch & 1][M];
      float cn1 = s_cnorm[ch & 1][16 + M];
      int col0 = ch * CHUNK + M;
      int col1 = col0 + 16;
      #pragma unroll
      for (int j = 0; j < 8; ++j) {
        float v0 = 2.0f * acc0[j] - cn0;
        if (v0 > rmax[j]) { rmax[j] = v0; ridx[j] = col0; }
        float v1 = 2.0f * acc1[j] - cn1;
        if (v1 > rmax[j]) { rmax[j] = v1; ridx[j] = col1; }
      }
    }

    wait_async0();     // chunk ch+1 landed (this wave)
    __syncthreads();   // workgroup-wide: compute(ch) done, chunk ch+1 visible
  }

  // ---- cross-lane argmax reduction (16 lanes per row), tie -> smaller index ----
  #pragma unroll
  for (int j = 0; j < 8; ++j) {
    float v  = rmax[j];
    int   ix = ridx[j];
    #pragma unroll
    for (int m = 1; m <= 8; m <<= 1) {
      float ov = __shfl_xor(v, m, 32);
      int   oi = __shfl_xor(ix, m, 32);
      if (ov > v || (ov == v && oi < ix)) { v = ov; ix = oi; }
    }
    if (M == 0) {                       // lanes 0 and 16 hold rows j and j+8
      s_idx[wave * 16 + j + kh * 8] = ix;
      atomicAdd(&wsi[WS_CNT + ix], 1); // vq_current_count scatter-add
    }
  }
  __syncthreads();

  // ---- write q (fp32 gather from codebook) + fused MSE reduction ----
  float lsum = 0.f;
  for (int i = 0; i < 128; ++i) {
    int flat = i * THREADS + tid;        // 0..16383 = 64 rows x 256 dims
    int r = flat & 63;                   // fastest -> coalesced q stores
    int d = flat >> 6;
    long long n = nBase + r;
    long long addr = (n >> 12) * ((long long)D_DIM * HW)
                   + (long long)d * HW + (n & 4095);
    float cv = cb[s_idx[r] * D_DIM + d];
    float zv = z[addr];
    float df = cv - zv;
    lsum += df * df;
    q[addr] = cv;                        // straight-through: q == z_quantized
  }
  atomicAdd(&s_acc, lsum);
  __syncthreads();
  if (tid == 0) atomicAdd(&wsf[WS_SQSUM], s_acc);
}

// ---------------- Kernel 3: scalars + counts into d_out tail ----------------
__global__ void vq_fin(const float* wsf, const int* wsi, float* out) {
  const long long qsize = (long long)N_TOT * D_DIM;
  int tid = threadIdx.x;
  if (tid == 0) {
    float mse = wsf[WS_SQSUM] / (float)((long long)N_TOT * D_DIM);
    out[qsize]     = mse * 1.25f;  // loss = codebook + BETA*commitment
    out[qsize + 1] = mse;          // codebook_loss
    out[qsize + 2] = mse;          // commitment_loss (same value in fwd)
  }
  for (int k = tid; k < KCODES; k += blockDim.x)
    out[qsize + 3 + k] = (float)wsi[WS_CNT + k];
}

extern "C" void kernel_launch(void* const* d_in, const int* in_sizes, int n_in,
                              void* d_out, int out_size, void* d_ws, size_t ws_size,
                              hipStream_t stream) {
  const float* z  = (const float*)d_in[0];   // (16,256,64,64) fp32
  const float* cb = (const float*)d_in[1];   // (2048,256) fp32
  float* out = (float*)d_out;
  float* wsf = (float*)d_ws;
  int*   wsi = (int*)d_ws;
  unsigned int* cbbf = (unsigned int*)d_ws + WS_CBBF;

  vq_prep<<<KCODES / 256, 256, 0, stream>>>(cb, wsf, wsi, cbbf);
  vq_main<<<N_TOT / ROWS_PB, THREADS, 0, stream>>>(z, cb, cbbf, out, wsf, wsi);
  vq_fin<<<1, 256, 0, stream>>>(wsf, wsi, out);
}
